// BiLSTM_CRF_41961830482500
// MI455X (gfx1250) — matrix operand order repaired and checked by reading, not compiled, not run on hardware
//
#include <hip/hip_runtime.h>
#include <hip/hip_bf16.h>
#include <math.h>

#define VOCABN 50000
#define EMBD   256
#define HIDD   512
#define NTAGS  32
#define NLAYER 3
#define BBATCH 32
#define SEQ    256
#define HH     256     // HIDD/2 per direction
#define G4     1024    // 4*HH gate width
#define BS     8192    // BBATCH*SEQ

typedef __bf16 bf16x16 __attribute__((ext_vector_type(16)));
typedef float  f32x8   __attribute__((ext_vector_type(8)));

union Frag { bf16x16 v; unsigned u[8]; };

__device__ __forceinline__ unsigned short f2bf(float f) {
  unsigned u = __float_as_uint(f);
  u += 0x7FFFu + ((u >> 16) & 1u);          // round-to-nearest-even
  return (unsigned short)(u >> 16);
}
__device__ __forceinline__ float sigf(float x) { return 1.0f / (1.0f + __expf(-x)); }

// ---------------------------------------------------------------------------
// CDNA5 async global->LDS copy (ASYNCcnt-tracked).
//   void __builtin_amdgcn_global_load_async_to_lds_b128(
//       int4 addrspace(1)* gsrc, int4 addrspace(3)* ldsdst, imm offset, imm cpol)
// ---------------------------------------------------------------------------
#if __has_builtin(__builtin_amdgcn_global_load_async_to_lds_b128) && \
    __has_builtin(__builtin_amdgcn_s_wait_asynccnt)
#define HAVE_ASYNC_LDS 1
typedef int v4i_t __attribute__((vector_size(16)));
typedef __attribute__((address_space(1))) v4i_t* gptr128_t;
typedef __attribute__((address_space(3))) v4i_t* lptr128_t;
__device__ __forceinline__ gptr128_t as_global(const void* p) {
  return (gptr128_t)(unsigned long long)p;          // generic global == as(1) bits
}
__device__ __forceinline__ lptr128_t as_lds(void* p) {
  return (lptr128_t)(unsigned)(unsigned long long)p; // LDS offset = low 32 bits
}
#define ASYNC_CP16(gsrc, ldst) \
  __builtin_amdgcn_global_load_async_to_lds_b128(as_global(gsrc), as_lds(ldst), 0, 0)
#define ASYNC_WAIT_N(n) __builtin_amdgcn_s_wait_asynccnt(n)
#else
#define HAVE_ASYNC_LDS 0
#define ASYNC_WAIT_N(n)
#endif

// ---------------------------------------------------------------------------
// Embedding gather -> bf16 layer-0 input, rows r = t*B + b. float4 vectorized.
// ---------------------------------------------------------------------------
__global__ void embed_kernel(const int* __restrict__ x, const float* __restrict__ emb,
                             unsigned short* __restrict__ X) {
  const int idx = blockIdx.x * 256 + threadIdx.x;   // BS*EMBD/4 chunks
  const int r   = idx >> 6;                         // EMBD/4 = 64 chunks per row
  const int c4  = (idx & 63) * 4;
  const int t   = r >> 5;
  const int b   = r & 31;
  const int tok = x[b * SEQ + t];
  const float4 v = *(const float4*)(emb + (size_t)tok * EMBD + c4);
  ushort4 o;
  o.x = f2bf(v.x); o.y = f2bf(v.y); o.z = f2bf(v.z); o.w = f2bf(v.w);
  *(ushort4*)(X + (size_t)r * EMBD + c4) = o;
}

// f32 -> bf16 row-major weight conversion with optional zero row-padding.
// nchunks = npad_rows * K / 4; rows >= nrows are written as zeros.
__global__ void cvt_pad_kernel(const float* __restrict__ in, unsigned short* __restrict__ out,
                               int nrows, int K, int nchunks) {
  const int c = blockIdx.x * blockDim.x + threadIdx.x;
  if (c >= nchunks) return;
  const int kc  = K >> 2;
  const int row = c / kc;
  const int kq  = (c - row * kc) * 4;
  ushort4 o;
  if (row < nrows) {
    const float4 v = *(const float4*)(in + (size_t)row * K + kq);
    o.x = f2bf(v.x); o.y = f2bf(v.y); o.z = f2bf(v.z); o.w = f2bf(v.w);
  } else {
    o.x = 0; o.y = 0; o.z = 0; o.w = 0;
  }
  *(ushort4*)(out + (size_t)row * K + kq) = o;
}

// ---------------------------------------------------------------------------
// C[M][N] = A(bf16)[M][K] @ Wbf(bf16)[Npad][K]^T + bias0 + bias1
// Block: 128 threads (4 waves), 64x64 tile, K-step 32, bf16 WMMA f32-accum.
// Both tiles staged with async b128 copies into double-buffered LDS; the next
// K-slice is in flight while the current one feeds the WMMAs
// (s_wait_asynccnt 4 == older buffer resident; async loads retire in order).
// ---------------------------------------------------------------------------
__global__ void gemm_bf16_wmma(const unsigned short* __restrict__ A, int lda,
                               const unsigned short* __restrict__ Wbf, // [Npad][K]
                               const float* __restrict__ bias0,
                               const float* __restrict__ bias1,
                               float* __restrict__ C, int ldc,
                               int M, int N, int K) {
  __shared__ __align__(16) unsigned short a_s[2][64 * 32];   // [m][k]
  __shared__ __align__(16) unsigned short b_s[2][64 * 32];   // [n][k]
  const int tid  = threadIdx.x;
  const int wave = tid >> 5;
  const int lane = tid & 31;
  const int kg   = lane >> 4;     // lane half selects K-group (ISA 7.12.2)
  const int lr   = lane & 15;
  const int m0   = blockIdx.y * 64;
  const int n0   = blockIdx.x * 64;
  const int nb   = wave * 16;     // each wave owns one 16-wide N strip

  auto stage = [&](int k0, int buf) {
#if HAVE_ASYNC_LDS
    #pragma unroll
    for (int i = 0; i < 2; ++i) {             // A: 256 x 16B chunks
      const int cid = tid + i * 128;
      const int r = cid >> 2, kq = (cid & 3) * 8;
      ASYNC_CP16(A + (size_t)(m0 + r) * lda + k0 + kq, &a_s[buf][r * 32 + kq]);
    }
    #pragma unroll
    for (int i = 0; i < 2; ++i) {             // W: 256 x 16B chunks
      const int cid = tid + i * 128;
      const int n = cid >> 2, kq = (cid & 3) * 8;
      ASYNC_CP16(Wbf + (size_t)(n0 + n) * K + k0 + kq, &b_s[buf][n * 32 + kq]);
    }
#else
    #pragma unroll
    for (int i = 0; i < 4; ++i) {
      const int cid = tid + i * 128;
      const int r = cid >> 3, kq = (cid & 7) * 4;
      *(uint2*)&a_s[buf][r * 32 + kq] =
          *(const uint2*)(A + (size_t)(m0 + r) * lda + k0 + kq);
      *(uint2*)&b_s[buf][r * 32 + kq] =
          *(const uint2*)(Wbf + (size_t)(n0 + r) * K + k0 + kq);
    }
#endif
  };

  f32x8 acc[4] = {};
  const int nk = K >> 5;
  stage(0, 0);

  for (int ki = 0; ki < nk; ++ki) {
    const int cur = ki & 1;
    const bool more = (ki + 1 < nk);
    if (more) stage((ki + 1) << 5, 1 - cur);  // overlap next K-slice with compute
#if HAVE_ASYNC_LDS
    if (more) ASYNC_WAIT_N(4); else ASYNC_WAIT_N(0);
#endif
    __syncthreads();

    // B fragment: lane col N = lr, K = {kg*8..+7, 16+kg*8..+7}
    Frag bfr;
    const unsigned* bp = (const unsigned*)&b_s[cur][(nb + lr) * 32 + kg * 8];
    #pragma unroll
    for (int i = 0; i < 8; ++i) bfr.u[i] = bp[(i & 3) + (i >> 2) * 8];

    #pragma unroll
    for (int mi = 0; mi < 4; ++mi) {
      // A fragment: row M = lr, K = {kg*8..+7, 16+kg*8..+7}
      Frag afr;
      const unsigned* a0 = (const unsigned*)&a_s[cur][(mi * 16 + lr) * 32 + kg * 8];
      const unsigned* a1 = (const unsigned*)&a_s[cur][(mi * 16 + lr) * 32 + 16 + kg * 8];
      #pragma unroll
      for (int i = 0; i < 4; ++i) { afr.u[i] = a0[i]; afr.u[4 + i] = a1[i]; }
      acc[mi] = __builtin_amdgcn_wmma_f32_16x16x32_bf16(
          false, afr.v, false, bfr.v, (short)0, acc[mi], false, false);
    }
    __syncthreads();   // WAR: next stage() rewrites the other buffer
  }

  const int gn = n0 + nb + lr;
  if (gn < N) {
    float bv = 0.f;
    if (bias0) bv += bias0[gn];
    if (bias1) bv += bias1[gn];
    #pragma unroll
    for (int mi = 0; mi < 4; ++mi) {
      #pragma unroll
      for (int r = 0; r < 8; ++r) {            // C layout: VGPR r -> M = r + 8*kg
        const int row = m0 + mi * 16 + r + kg * 8;
        C[(size_t)row * ldc + gn] = acc[mi][r] + bv;
      }
    }
  }
}

// ---------------------------------------------------------------------------
// Recurrent scan: one workgroup (32 waves = full WGP) per direction.
// LDS: h (bf16, 16KB) + double-buffered preact tile (2 x 128KB).
// The preact tile for step t+1 is async-copied to LDS during step t's WMMAs,
// so the only recurrence-critical work is h@Whh^T (32 WMMAs) + the cell math.
// Gates are accumulated in place into the current preact LDS buffer.
// ---------------------------------------------------------------------------
__global__ void lstm_layer_kernel(const float* __restrict__ preF,
                                  const float* __restrict__ preB,
                                  const unsigned short* __restrict__ whhF,
                                  const unsigned short* __restrict__ whhB,
                                  unsigned short* __restrict__ Yout /* [BS][HIDD] bf16 */) {
  const int dir = blockIdx.x;
  const float* __restrict__ pre = dir ? preB : preF;
  const unsigned short* __restrict__ whh = dir ? whhB : whhF;

  extern __shared__ char smem[];
  unsigned short* hbf = (unsigned short*)smem;                          // 32*256 bf16
  float* pre_s[2];
  pre_s[0] = (float*)(smem + 32 * 256 * sizeof(unsigned short));        // 32*1024 f32
  pre_s[1] = pre_s[0] + 32 * 1024;                                      // 32*1024 f32

  const int tid  = threadIdx.x;
  const int wave = tid >> 5;
  const int lane = tid & 31;
  const int kg   = lane >> 4;
  const int lr   = lane & 15;
  const int wn   = wave * 32;     // wave owns a 32-wide N strip of the gate matrix

  auto stage_pre = [&](int tt, float* dst) {
#if HAVE_ASYNC_LDS
    #pragma unroll
    for (int i = 0; i < 8; ++i) {
      const int cid = tid + i * 1024;         // 8192 x 16B chunks (128KB)
      ASYNC_CP16(pre + (size_t)tt * (BBATCH * G4) + cid * 4, dst + cid * 4);
    }
#else
    #pragma unroll
    for (int i = 0; i < 8; ++i) {
      const int cid = tid + i * 1024;
      *(float4*)(dst + cid * 4) =
          *(const float4*)(pre + (size_t)tt * (BBATCH * G4) + cid * 4);
    }
#endif
  };

  for (int i = tid; i < 32 * 256; i += 1024) hbf[i] = 0;
  float c[8];
  #pragma unroll
  for (int i = 0; i < 8; ++i) c[i] = 0.f;

  const int t0 = dir ? (SEQ - 1) : 0;
  stage_pre(t0, pre_s[0]);

  for (int step = 0; step < SEQ; ++step) {
    const int t = dir ? (SEQ - 1 - step) : step;
    const int cur = step & 1;
    float* __restrict__ g_s = pre_s[cur];

    ASYNC_WAIT_N(0);       // pre_s[cur] in-flight copies (issued last step) done
    __syncthreads();       // ..visible to all waves; also orders hbf writes

    if (step + 1 < SEQ) {  // kick off next step's preact copy into other buffer
      const int tn = dir ? (t - 1) : (t + 1);
      stage_pre(tn, pre_s[1 - cur]);
    }

    // ---- gates += h @ Whh^T via WMMA (K = 256, 8 k-steps, 4 tiles/wave) ----
    f32x8 acc[2][2] = {};
    #pragma unroll
    for (int k0 = 0; k0 < HH; k0 += 32) {
      Frag bfr[2];
      #pragma unroll
      for (int ni = 0; ni < 2; ++ni) {
        const unsigned* bp =
            (const unsigned*)(whh + (size_t)(wn + ni * 16 + lr) * HH + k0 + kg * 8);
        #pragma unroll
        for (int i = 0; i < 8; ++i) bfr[ni].u[i] = bp[(i & 3) + (i >> 2) * 8];
      }
      Frag afr[2];
      #pragma unroll
      for (int mi = 0; mi < 2; ++mi) {
        const unsigned* a0 = (const unsigned*)(hbf + (mi * 16 + lr) * HH + k0 + kg * 8);
        const unsigned* a1 = (const unsigned*)(hbf + (mi * 16 + lr) * HH + k0 + 16 + kg * 8);
        #pragma unroll
        for (int i = 0; i < 4; ++i) { afr[mi].u[i] = a0[i]; afr[mi].u[4 + i] = a1[i]; }
      }
      #pragma unroll
      for (int mi = 0; mi < 2; ++mi)
        #pragma unroll
        for (int ni = 0; ni < 2; ++ni)
          acc[mi][ni] = __builtin_amdgcn_wmma_f32_16x16x32_bf16(
              false, afr[mi].v, false, bfr[ni].v, (short)0, acc[mi][ni], false, false);
    }

    // accumulate gates in place into the LDS preact tile
    #pragma unroll
    for (int mi = 0; mi < 2; ++mi)
      #pragma unroll
      for (int ni = 0; ni < 2; ++ni) {
        const int gcol = wn + ni * 16 + lr;
        #pragma unroll
        for (int r = 0; r < 8; ++r) {
          const int row = mi * 16 + r + kg * 8;
          g_s[row * G4 + gcol] += acc[mi][ni][r];
        }
      }
    __syncthreads();

    // ---- elementwise LSTM cell; h -> LDS (bf16) and layer output ----
    #pragma unroll
    for (int i = 0; i < 8; ++i) {
      const int flat = tid + i * 1024;
      const int m = flat >> 8;
      const int q = flat & 255;
      const float gi = g_s[m * G4 + q];
      const float gf = g_s[m * G4 + 256 + q];
      const float gg = g_s[m * G4 + 512 + q];
      const float go = g_s[m * G4 + 768 + q];
      c[i] = sigf(gf) * c[i] + sigf(gi) * tanhf(gg);
      const float h = sigf(go) * tanhf(c[i]);
      const unsigned short hb = f2bf(h);
      hbf[m * HH + q] = hb;
      Yout[(size_t)(t * BBATCH + m) * HIDD + dir * HH + q] = hb;
    }
    // next iteration's top-of-loop barrier orders hbf for the next GEMM
  }
}

// ---------------------------------------------------------------------------
// CRF: gold-path score + forward-algorithm logsumexp scan.
// One wave per batch element, lane = tag.
// ---------------------------------------------------------------------------
__global__ void crf_kernel(const float* __restrict__ emis,   // [(t*B+b)][32]
                           const int* __restrict__ x,
                           const int* __restrict__ tags,
                           const float* __restrict__ trans,
                           const float* __restrict__ startt,
                           const float* __restrict__ endt,
                           float* __restrict__ out) {
  __shared__ float trans_s[NTAGS * NTAGS];
  __shared__ float red[BBATCH];
  const int tid = threadIdx.x;
  const int b = tid >> 5;
  const int j = tid & 31;
  if (tid < NTAGS * NTAGS) trans_s[tid] = trans[tid];
  __syncthreads();

  // gold score, time-strided over lanes then wave-reduced
  float sc = 0.f;
  for (int t = j; t < SEQ; t += 32) {
    const int tg = tags[b * SEQ + t];
    const float mt = (x[b * SEQ + t] != 0) ? 1.f : 0.f;
    sc += emis[((size_t)(t * BBATCH + b) << 5) + tg] * mt;
    if (t >= 1) sc += trans_s[tags[b * SEQ + t - 1] * NTAGS + tg] * mt;
  }
  if (j == 0) sc += startt[tags[b * SEQ + 0]];
  if (j == 31) {
    const float ml = (x[b * SEQ + SEQ - 1] != 0) ? 1.f : 0.f;
    sc += endt[tags[b * SEQ + SEQ - 1]] * ml;
  }
  #pragma unroll
  for (int off = 16; off >= 1; off >>= 1) sc += __shfl_xor(sc, off, 32);

  // forward algorithm (mask term is identity in the forward value)
  float la = startt[j] + emis[((size_t)b << 5) + j];
  for (int t = 1; t < SEQ; ++t) {
    const float em = emis[((size_t)(t * BBATCH + b) << 5) + j];
    float mx = -3.0e38f;
    #pragma unroll 4
    for (int i = 0; i < 32; ++i) {
      const float v = __shfl(la, i, 32) + trans_s[i * NTAGS + j];
      mx = fmaxf(mx, v);
    }
    float se = 0.f;
    #pragma unroll 4
    for (int i = 0; i < 32; ++i) {
      const float v = __shfl(la, i, 32) + trans_s[i * NTAGS + j];
      se += __expf(v - mx);
    }
    la = mx + __logf(se) + em;
  }
  float v = la + endt[j];
  float mx = v;
  #pragma unroll
  for (int off = 16; off >= 1; off >>= 1) mx = fmaxf(mx, __shfl_xor(mx, off, 32));
  float se = __expf(v - mx);
  #pragma unroll
  for (int off = 16; off >= 1; off >>= 1) se += __shfl_xor(se, off, 32);
  const float lz = mx + __logf(se);

  if (j == 0) red[b] = lz - sc;
  __syncthreads();
  if (tid == 0) {
    float s = 0.f;
    for (int i = 0; i < BBATCH; ++i) s += red[i];
    out[0] = s / (float)BBATCH;
  }
}

// ---------------------------------------------------------------------------
extern "C" void kernel_launch(void* const* d_in, const int* in_sizes, int n_in,
                              void* d_out, int out_size, void* d_ws, size_t ws_size,
                              hipStream_t stream) {
  (void)in_sizes; (void)n_in; (void)out_size; (void)ws_size;
  const int*   x     = (const int*)d_in[0];
  const int*   tags  = (const int*)d_in[1];
  const float* embed = (const float*)d_in[2];
  auto LP = [&](int l, int d, int p) -> const float* {
    return (const float*)d_in[3 + (l * 2 + d) * 4 + p];   // p: 0=Wih 1=Whh 2=bih 3=bhh
  };
  const float* W_out  = (const float*)d_in[27];
  const float* b_out  = (const float*)d_in[28];
  const float* trans  = (const float*)d_in[29];
  const float* startt = (const float*)d_in[30];
  const float* endt   = (const float*)d_in[31];
  float* out = (float*)d_out;

  char* w = (char*)d_ws;
  size_t off = 0;
  auto wsalloc = [&](size_t bytes) -> void* {
    void* p = w + off;
    off = (off + bytes + 255) & ~(size_t)255;
    return p;
  };
  unsigned short* Xbf    = (unsigned short*)wsalloc((size_t)BS * HIDD * 2);
  unsigned short* Ybf    = (unsigned short*)wsalloc((size_t)BS * HIDD * 2);
  float*          preF   = (float*)wsalloc((size_t)SEQ * BBATCH * G4 * 4);
  float*          preB   = (float*)wsalloc((size_t)SEQ * BBATCH * G4 * 4);
  unsigned short* wihF   = (unsigned short*)wsalloc((size_t)G4 * HIDD * 2);
  unsigned short* wihB   = (unsigned short*)wsalloc((size_t)G4 * HIDD * 2);
  unsigned short* whhF   = (unsigned short*)wsalloc((size_t)G4 * HH * 2);
  unsigned short* whhB   = (unsigned short*)wsalloc((size_t)G4 * HH * 2);
  unsigned short* woutbf = (unsigned short*)wsalloc((size_t)64 * HIDD * 2);
  float*          emis   = (float*)wsalloc((size_t)BS * NTAGS * 4);

  embed_kernel<<<(BS * EMBD / 4) / 256, 256, 0, stream>>>(x, embed, Xbf);

  const size_t lstm_smem =
      32 * 256 * sizeof(unsigned short) + 2 * 32 * 1024 * sizeof(float);  // 272KB
  for (int l = 0; l < NLAYER; ++l) {
    const int in_dim = (l == 0) ? EMBD : HIDD;
    unsigned short* IN  = (l % 2 == 0) ? Xbf : Ybf;
    unsigned short* OUT = (l % 2 == 0) ? Ybf : Xbf;
    const int wih_chunks = G4 * in_dim / 4;
    const int whh_chunks = G4 * HH / 4;
    cvt_pad_kernel<<<(wih_chunks + 255) / 256, 256, 0, stream>>>(LP(l, 0, 0), wihF, G4, in_dim, wih_chunks);
    cvt_pad_kernel<<<(wih_chunks + 255) / 256, 256, 0, stream>>>(LP(l, 1, 0), wihB, G4, in_dim, wih_chunks);
    cvt_pad_kernel<<<(whh_chunks + 255) / 256, 256, 0, stream>>>(LP(l, 0, 1), whhF, G4, HH, whh_chunks);
    cvt_pad_kernel<<<(whh_chunks + 255) / 256, 256, 0, stream>>>(LP(l, 1, 1), whhB, G4, HH, whh_chunks);
    dim3 g(G4 / 64, BS / 64);
    gemm_bf16_wmma<<<g, 128, 0, stream>>>(IN, in_dim, wihF, LP(l, 0, 2), LP(l, 0, 3),
                                          preF, G4, BS, G4, in_dim);
    gemm_bf16_wmma<<<g, 128, 0, stream>>>(IN, in_dim, wihB, LP(l, 1, 2), LP(l, 1, 3),
                                          preB, G4, BS, G4, in_dim);
    lstm_layer_kernel<<<2, 1024, lstm_smem, stream>>>(preF, preB, whhF, whhB, OUT);
  }

  // emissions: final bidirectional features live in Ybf (layer 2 output);
  // W_out converted once into a 64-row zero-padded bf16 tile.
  const int wout_chunks = 64 * HIDD / 4;
  cvt_pad_kernel<<<(wout_chunks + 255) / 256, 256, 0, stream>>>(W_out, woutbf, NTAGS, HIDD, wout_chunks);
  gemm_bf16_wmma<<<dim3(1, BS / 64), 128, 0, stream>>>(Ybf, HIDD, woutbf, b_out, nullptr,
                                                       emis, NTAGS, BS, NTAGS, HIDD);
  crf_kernel<<<1, 1024, 0, stream>>>(emis, x, tags, trans, startt, endt, out);
}
